// MaskedMultiHeadAttention_14654428413998
// MI455X (gfx1250) — matrix operand order repaired and checked by reading, not compiled, not run on hardware
//
#include <hip/hip_runtime.h>
#include <hip/hip_bf16.h>

// ---- fixed problem dims ----
constexpr int D_MODEL = 1024;
constexpr int HEADS   = 16;
constexpr int HD      = 64;
constexpr int SEQ     = 2048;
constexpr int BATCH   = 2;
constexpr int M_TOT   = BATCH * SEQ;   // 4096

typedef _Float16 v16h __attribute__((ext_vector_type(16)));
typedef _Float16 v8h  __attribute__((ext_vector_type(8)));
typedef _Float16 v4h  __attribute__((ext_vector_type(4)));
typedef float    v8f  __attribute__((ext_vector_type(8)));
typedef float    v4f  __attribute__((ext_vector_type(4)));

// =====================================================================
// Transpose + convert to f16:  dst[c][r] = (f16) src[r][c]
// src is [R][C] row-major (per blockIdx.z slab), dst is [C][R] row-major.
// 64x64 tiles through LDS (stride 72 to avoid bank conflicts).
// =====================================================================
template <typename ST>
__global__ __launch_bounds__(256) void transpose_f16_kernel(
    const ST* __restrict__ src, _Float16* __restrict__ dst, int R, int C) {
  __shared__ alignas(16) _Float16 tile[64 * 72];
  const int t = threadIdx.x;
  const size_t zoff = (size_t)blockIdx.z * R * C;
  src += zoff;
  dst += zoff;
  const int r0 = blockIdx.y * 64, c0 = blockIdx.x * 64;

  {
    const int row = t >> 2;           // 0..63
    const int col = (t & 3) * 16;     // 0,16,32,48
    if constexpr (sizeof(ST) == 4) {
#pragma unroll
      for (int j = 0; j < 4; ++j) {
        v4f v = *(const v4f*)(src + (size_t)(r0 + row) * C + c0 + col + j * 4);
        *(v4h*)&tile[row * 72 + col + j * 4] = __builtin_convertvector(v, v4h);
      }
    } else {
#pragma unroll
      for (int j = 0; j < 2; ++j)
        *(v8h*)&tile[row * 72 + col + j * 8] =
            *(const v8h*)(src + (size_t)(r0 + row) * C + c0 + col + j * 8);
    }
  }
  __syncthreads();
  {
    const int cc = t >> 2;            // dst row (src column)
    const int sr = (t & 3) * 16;      // src row start
    v8h o0, o1;
#pragma unroll
    for (int j = 0; j < 8; ++j) o0[j] = tile[(sr + j) * 72 + cc];
#pragma unroll
    for (int j = 0; j < 8; ++j) o1[j] = tile[(sr + 8 + j) * 72 + cc];
    _Float16* dp = dst + (size_t)(c0 + cc) * R + r0 + sr;
    *(v8h*)dp = o0;
    *(v8h*)(dp + 8) = o1;
  }
}

// =====================================================================
// WMMA GEMM: C[M x 1024] = (A[M x 1024] @ (BT)^T + bias) * scale
//   A: fp32 or f16 row-major, BT: f16 [N][K] row-major (pre-transposed)
//   MODE 0: out = fp32 row-major [M][N]
//   MODE 1: out = f16 per-head layout [B][H][S][HD]
// Block = 128x128 tile, 8 waves of 32x64, KSTEP=32.
// Double-buffered LDS (reg->LDS pipeline, one barrier per k-step) +
// global_prefetch two tiles ahead.
// =====================================================================
template <typename AT, int MODE>
__global__ __launch_bounds__(256) void gemm_wmma_kernel(
    const AT* __restrict__ A, const _Float16* __restrict__ BT,
    const float* __restrict__ bias, float scale, void* __restrict__ outp) {
  constexpr int K = 1024, N = 1024;
  constexpr int LDA = 40;  // padded f16 stride (80B = 20 banks -> conflict-free)
  __shared__ alignas(16) _Float16 sA[2][128 * LDA];
  __shared__ alignas(16) _Float16 sB[2][128 * LDA];

  const int t = threadIdx.x;
  const int m0 = blockIdx.y * 128;
  const int n0 = blockIdx.x * 128;
  const int w = t >> 5, l = t & 31, lq = l & 15, half = l >> 4;
  const int wm = w >> 1, wn = w & 1;

  v8f c[2][4] = {};

  // staging registers
  v4h ar4[4];  // AT == float path
  v8h ar8[2];  // AT == f16 path
  v8h br[2];

  auto load_tiles = [&](int kt) {
    if constexpr (sizeof(AT) == 4) {
#pragma unroll
      for (int p = 0; p < 4; ++p) {
        const int row = p * 32 + (t >> 3);
        const int col = (t & 7) * 4;
        v4f av = *(const v4f*)(A + (size_t)(m0 + row) * K + kt + col);
        ar4[p] = __builtin_convertvector(av, v4h);
      }
    } else {
#pragma unroll
      for (int p = 0; p < 2; ++p) {
        const int row = p * 64 + (t >> 2);
        const int col = (t & 3) * 8;
        ar8[p] = *(const v8h*)(A + (size_t)(m0 + row) * K + kt + col);
      }
    }
#pragma unroll
    for (int p = 0; p < 2; ++p) {
      const int row = p * 64 + (t >> 2);
      const int col = (t & 3) * 8;
      br[p] = *(const v8h*)(BT + (size_t)(n0 + row) * K + kt + col);
    }
    // prefetch two tiles ahead (gfx1250 global_prefetch_b8 path)
    if (kt + 64 < K) {
      __builtin_prefetch(A + (size_t)(m0 + (t >> 1)) * K + kt + 64, 0, 1);
      __builtin_prefetch(BT + (size_t)(n0 + (t >> 1)) * K + kt + 64, 0, 1);
    }
  };

  auto store_tiles = [&](int buf) {
    if constexpr (sizeof(AT) == 4) {
#pragma unroll
      for (int p = 0; p < 4; ++p) {
        const int row = p * 32 + (t >> 3);
        const int col = (t & 7) * 4;
        *(v4h*)&sA[buf][row * LDA + col] = ar4[p];
      }
    } else {
#pragma unroll
      for (int p = 0; p < 2; ++p) {
        const int row = p * 64 + (t >> 2);
        const int col = (t & 3) * 8;
        *(v8h*)&sA[buf][row * LDA + col] = ar8[p];
      }
    }
#pragma unroll
    for (int p = 0; p < 2; ++p) {
      const int row = p * 64 + (t >> 2);
      const int col = (t & 3) * 8;
      *(v8h*)&sB[buf][row * LDA + col] = br[p];
    }
  };

  load_tiles(0);
  int buf = 0;
  for (int kt = 0; kt < K; kt += 32) {
    store_tiles(buf);
    __syncthreads();
    if (kt + 32 < K) load_tiles(kt + 32);  // overlap next tile with WMMAs

    v16h af[2], bf[4];
#pragma unroll
    for (int mf = 0; mf < 2; ++mf) {
      // A layout: lane half 0 -> K{0..7,16..23}, half 1 -> K{8..15,24..31}
      const _Float16* pr = &sA[buf][(wm * 32 + mf * 16 + lq) * LDA + half * 8];
      *(v8h*)&af[mf] = *(const v8h*)pr;
      *((v8h*)&af[mf] + 1) = *(const v8h*)(pr + 16);
    }
#pragma unroll
    for (int nf = 0; nf < 4; ++nf) {
      // B layout: lane half 0 -> K{0..7,16..23}, half 1 -> K{8..15,24..31}
      const _Float16* pr = &sB[buf][(wn * 64 + nf * 16 + lq) * LDA + half * 8];
      *(v8h*)&bf[nf] = *(const v8h*)pr;
      *((v8h*)&bf[nf] + 1) = *(const v8h*)(pr + 16);
    }
#pragma unroll
    for (int mf = 0; mf < 2; ++mf)
#pragma unroll
      for (int nf = 0; nf < 4; ++nf)
        c[mf][nf] = __builtin_amdgcn_wmma_f32_16x16x32_f16(
            false, af[mf], false, bf[nf], (short)0, c[mf][nf], false, false);
    buf ^= 1;
  }

  // ---- epilogue: bias add (+scale) + store ----
#pragma unroll
  for (int mf = 0; mf < 2; ++mf) {
#pragma unroll
    for (int nf = 0; nf < 4; ++nf) {
      const int col = n0 + wn * 64 + nf * 16 + lq;
      const float bv = bias[col];
#pragma unroll
      for (int r = 0; r < 8; ++r) {
        const int row = m0 + wm * 32 + mf * 16 + half * 8 + r;
        const float val = (c[mf][nf][r] + bv) * scale;
        if constexpr (MODE == 0) {
          ((float*)outp)[(size_t)row * N + col] = val;
        } else {
          const int b = row >> 11, s = row & (SEQ - 1);
          const int h = col >> 6, d = col & (HD - 1);
          ((_Float16*)outp)[((size_t)(b * HEADS + h) * SEQ + s) * HD + d] =
              (_Float16)val;
        }
      }
    }
  }
}

// =====================================================================
// Flash attention (causal). One (b,h) per blockIdx.y, 128 queries/block,
// wave-private 16-query tiles, 64-key tiles. K and V^T tiles are staged
// ONCE per block into double-buffered LDS (8x less L2 traffic than
// per-wave fragment loads). Trip count is block-uniform; waves whose
// tile is fully causally masked skip the compute (but still stage+sync).
// Q is pre-scaled by 1/sqrt(HD) in the projection epilogue.
//   qh/kh: [B*H][SEQ][HD] f16      vT: [B*H][HD][SEQ] f16
//   ctx:   [B][SEQ][D_MODEL] f16
// =====================================================================
__global__ __launch_bounds__(256) void attn_kernel(
    const _Float16* __restrict__ qh, const _Float16* __restrict__ kh,
    const _Float16* __restrict__ vT, _Float16* __restrict__ ctx) {
  __shared__ alignas(16) _Float16 sK[2][64 * 72];  // [key][dim]
  __shared__ alignas(16) _Float16 sV[2][64 * 72];  // [dim][key]
  __shared__ alignas(16) _Float16 sp[8 * 16 * 72]; // per-wave P strip
  const int t = threadIdx.x;
  const int w = t >> 5, l = t & 31, lq = l & 15, half = l >> 4;
  const int bh = blockIdx.y;
  const int q0 = blockIdx.x * 128 + w * 16;
  const int qmax = q0 + 15;

  const _Float16* qp = qh + (size_t)bh * SEQ * HD;
  const _Float16* kp = kh + (size_t)bh * SEQ * HD;
  const _Float16* vp = vT + (size_t)bh * HD * SEQ;
  _Float16* pw = &sp[w * 16 * 72];

  // resident Q fragments (A layout), K-dim = HD = 2 chunks of 32
  v16h qf[2];
#pragma unroll
  for (int kc = 0; kc < 2; ++kc) {
    const _Float16* p = qp + (size_t)(q0 + lq) * HD + kc * 32 + half * 8;
    *(v8h*)&qf[kc] = *(const v8h*)p;
    *((v8h*)&qf[kc] + 1) = *(const v8h*)(p + 16);
  }

  v8f acc[4] = {};
  float mst[8], lst[8];
#pragma unroll
  for (int r = 0; r < 8; ++r) { mst[r] = -1e30f; lst[r] = 0.f; }

  // cooperative staging pattern: each thread owns one row, 16 f16
  const int srow = t >> 2;          // 0..63
  const int scol = (t & 3) * 16;    // 0,16,32,48
  v8h kr[2], vr[2];
  auto load_tiles = [&](int kt) {
    const int kbase = kt * 64;
    const _Float16* kq = kp + (size_t)(kbase + srow) * HD + scol;
    kr[0] = *(const v8h*)kq;
    kr[1] = *(const v8h*)(kq + 8);
    const _Float16* vq = vp + (size_t)srow * SEQ + kbase + scol;
    vr[0] = *(const v8h*)vq;
    vr[1] = *(const v8h*)(vq + 8);
  };
  auto store_tiles = [&](int buf) {
    *(v8h*)&sK[buf][srow * 72 + scol]     = kr[0];
    *(v8h*)&sK[buf][srow * 72 + scol + 8] = kr[1];
    *(v8h*)&sV[buf][srow * 72 + scol]     = vr[0];
    *(v8h*)&sV[buf][srow * 72 + scol + 8] = vr[1];
  };

  const int nkt = 2 * blockIdx.x + 2;  // covers keys 0 .. blockQmax
  load_tiles(0);
  int buf = 0;
  for (int kt = 0; kt < nkt; ++kt) {
    const int kbase = kt * 64;
    store_tiles(buf);
    __syncthreads();
    if (kt + 1 < nkt) load_tiles(kt + 1);  // overlap with compute

    if (kbase <= qmax) {  // wave-uniform: skip fully-masked tiles
      // ---- S = Q K^T, B-operand rows straight from LDS K tile ----
      v8f sc[4] = {};
#pragma unroll
      for (int nf = 0; nf < 4; ++nf) {
#pragma unroll
        for (int kc = 0; kc < 2; ++kc) {
          v16h bf;
          const _Float16* p = &sK[buf][(nf * 16 + lq) * 72 + kc * 32 + half * 8];
          *(v8h*)&bf = *(const v8h*)p;
          *((v8h*)&bf + 1) = *(const v8h*)(p + 16);
          sc[nf] = __builtin_amdgcn_wmma_f32_16x16x32_f16(
              false, qf[kc], false, bf, (short)0, sc[nf], false, false);
        }
      }
      const bool need_mask = (kbase + 63) > q0;
      // ---- online softmax per row (rows live in lane halves) ----
#pragma unroll
      for (int r = 0; r < 8; ++r) {
        const int qrow = q0 + half * 8 + r;
        float rm = -1e30f;
#pragma unroll
        for (int nf = 0; nf < 4; ++nf) {
          float s = sc[nf][r];  // scale folded into Q projection
          if (need_mask && (kbase + nf * 16 + lq) > qrow) s = -1e30f;
          sc[nf][r] = s;
          rm = fmaxf(rm, s);
        }
        rm = fmaxf(rm, __shfl_xor(rm, 1, 32));
        rm = fmaxf(rm, __shfl_xor(rm, 2, 32));
        rm = fmaxf(rm, __shfl_xor(rm, 4, 32));
        rm = fmaxf(rm, __shfl_xor(rm, 8, 32));
        const float mnew = fmaxf(mst[r], rm);
        const float alpha = __expf(mst[r] - mnew);
        mst[r] = mnew;
        float rs = 0.f;
#pragma unroll
        for (int nf = 0; nf < 4; ++nf) {
          const float pv = __expf(sc[nf][r] - mnew);
          rs += pv;
          pw[(half * 8 + r) * 72 + nf * 16 + lq] = (_Float16)pv;  // D -> LDS
          acc[nf][r] *= alpha;
        }
        rs += __shfl_xor(rs, 1, 32);
        rs += __shfl_xor(rs, 2, 32);
        rs += __shfl_xor(rs, 4, 32);
        rs += __shfl_xor(rs, 8, 32);
        lst[r] = lst[r] * alpha + rs;
      }
      // DS ops are in-order per wave; wait + compiler barrier before re-read
      asm volatile("s_wait_dscnt 0x0" ::: "memory");
      // ---- O += P V : P re-read in A layout, V^T tile as B operand ----
#pragma unroll
      for (int kc = 0; kc < 2; ++kc) {
        v16h pf;
        const _Float16* pp = pw + lq * 72 + kc * 32 + half * 8;
        *(v8h*)&pf = *(const v8h*)pp;
        *((v8h*)&pf + 1) = *(const v8h*)(pp + 16);
#pragma unroll
        for (int nf = 0; nf < 4; ++nf) {
          v16h vf;
          const _Float16* vq = &sV[buf][(nf * 16 + lq) * 72 + kc * 32 + half * 8];
          *(v8h*)&vf = *(const v8h*)vq;
          *((v8h*)&vf + 1) = *(const v8h*)(vq + 16);
          acc[nf] = __builtin_amdgcn_wmma_f32_16x16x32_f16(
              false, pf, false, vf, (short)0, acc[nf], false, false);
        }
      }
      asm volatile("" ::: "memory");  // keep next P-stores after these reads
    }
    buf ^= 1;
  }

  // ---- normalize + store context in [B][SEQ][D_MODEL] f16 ----
  const int b = bh >> 4, h = bh & 15;
#pragma unroll
  for (int r = 0; r < 8; ++r) {
    const int qrow = q0 + half * 8 + r;
    const float inv = 1.0f / lst[r];
#pragma unroll
    for (int nf = 0; nf < 4; ++nf) {
      ctx[((size_t)(b * SEQ + qrow)) * D_MODEL + h * HD + nf * 16 + lq] =
          (_Float16)(acc[nf][r] * inv);
    }
  }
}

// =====================================================================
// Host launcher
// =====================================================================
extern "C" void kernel_launch(void* const* d_in, const int* in_sizes, int n_in,
                              void* d_out, int out_size, void* d_ws,
                              size_t ws_size, hipStream_t stream) {
  (void)in_sizes; (void)n_in; (void)out_size; (void)ws_size;
  const float* Q  = (const float*)d_in[0];
  const float* K  = (const float*)d_in[1];
  const float* V  = (const float*)d_in[2];
  // d_in[3] = causal mask (computed analytically in-kernel)
  const float* Wq = (const float*)d_in[4];
  const float* bq = (const float*)d_in[5];
  const float* Wk = (const float*)d_in[6];
  const float* bk = (const float*)d_in[7];
  const float* Wv = (const float*)d_in[8];
  const float* bv = (const float*)d_in[9];
  const float* Wo = (const float*)d_in[10];
  const float* bo = (const float*)d_in[11];

  char* ws = (char*)d_ws;
  size_t off = 0;
  auto alloc = [&](size_t bytes) -> char* {
    char* p = ws + off;
    off += (bytes + 255) & ~(size_t)255;
    return p;
  };
  const size_t WBYTES = (size_t)D_MODEL * D_MODEL * sizeof(_Float16);  // 2 MB
  const size_t ABYTES = (size_t)M_TOT * D_MODEL * sizeof(_Float16);    // 8 MB
  _Float16* WqT = (_Float16*)alloc(WBYTES);
  _Float16* WkT = (_Float16*)alloc(WBYTES);
  _Float16* WvT = (_Float16*)alloc(WBYTES);
  _Float16* WoT = (_Float16*)alloc(WBYTES);
  _Float16* qhB = (_Float16*)alloc(ABYTES);  // [B*H][S][HD]
  _Float16* khB = (_Float16*)alloc(ABYTES);
  _Float16* vhB = (_Float16*)alloc(ABYTES);
  _Float16* vTB = (_Float16*)alloc(ABYTES);  // [B*H][HD][S]
  _Float16* ctx = (_Float16*)alloc(ABYTES);  // [B][S][D]

  // 1) weights: fp32 [K][N] -> f16 [N][K]
  dim3 gW(D_MODEL / 64, D_MODEL / 64, 1);
  transpose_f16_kernel<float><<<gW, 256, 0, stream>>>(Wq, WqT, D_MODEL, D_MODEL);
  transpose_f16_kernel<float><<<gW, 256, 0, stream>>>(Wk, WkT, D_MODEL, D_MODEL);
  transpose_f16_kernel<float><<<gW, 256, 0, stream>>>(Wv, WvT, D_MODEL, D_MODEL);
  transpose_f16_kernel<float><<<gW, 256, 0, stream>>>(Wo, WoT, D_MODEL, D_MODEL);

  // 2) QKV projections -> per-head f16 layouts (Q pre-scaled by 1/sqrt(HD))
  dim3 gG(D_MODEL / 128, M_TOT / 128);
  gemm_wmma_kernel<float, 1><<<gG, 256, 0, stream>>>(Q, WqT, bq, 0.125f, qhB);
  gemm_wmma_kernel<float, 1><<<gG, 256, 0, stream>>>(K, WkT, bk, 1.0f, khB);
  gemm_wmma_kernel<float, 1><<<gG, 256, 0, stream>>>(V, WvT, bv, 1.0f, vhB);

  // 3) V -> V^T per head (for P@V B-operand)
  dim3 gV(HD / 64, SEQ / 64, BATCH * HEADS);
  transpose_f16_kernel<_Float16><<<gV, 256, 0, stream>>>(vhB, vTB, SEQ, HD);

  // 4) causal flash attention
  dim3 gA(SEQ / 128, BATCH * HEADS);
  attn_kernel<<<gA, 256, 0, stream>>>(qhB, khB, vTB, ctx);

  // 5) output projection -> fp32 d_out
  gemm_wmma_kernel<_Float16, 0><<<gG, 256, 0, stream>>>(ctx, WoT, bo, 1.0f,
                                                        d_out);
}
